// TransformerEncoder_78864189489908
// MI455X (gfx1250) — compile-verified, matrix operand order checked by reading, not run on hardware
//
#include <hip/hip_runtime.h>
#include <hip/hip_bf16.h>
#include <math.h>

// Problem constants (from reference)
#define BB 2
#define NN 2048
#define HH 16
#define HD 64
#define FF 1024
#define NT16 (NN / 16)          // 128 token tiles per batch row
#define SOFT_SCALE 0.03125f     // 1/sqrt(F) = 1/32

typedef __attribute__((ext_vector_type(16))) _Float16 v16h;
typedef __attribute__((ext_vector_type(8)))  _Float16 v8h;
typedef __attribute__((ext_vector_type(8)))  float    v8f;

__device__ __forceinline__ v8f wmma_ff(v16h a, v16h b, v8f c) {
  // D = A(16x32 f16) * B(32x16 f16) + C(16x16 f32)
  return __builtin_amdgcn_wmma_f32_16x16x32_f16(false, a, false, b, (short)0, c,
                                                false, false);
}

// Load a 16x32 f16 fragment from a row-major matrix with leading dim `ld`.
// Per CDNA5 ISA 16-bit A layout: lane L holds row (L&15); elements 0..7 are
// K = 8*(L>>4)+0..7, elements 8..15 are K = 16 + 8*(L>>4)+0..7.
// The same pattern loading X^T (row-major, lane = column) yields B = X.
__device__ __forceinline__ v16h frag_ld(const _Float16* p, int ld) {
  const int lane = threadIdx.x & 31;
  const _Float16* r = p + (size_t)(lane & 15) * ld + ((lane >> 4) << 3);
  v8h lo = *(const v8h*)(r);
  v8h hi = *(const v8h*)(r + 16);
  v16h f;
#pragma unroll
  for (int i = 0; i < 8; ++i) { f[i] = lo[i]; f[8 + i] = hi[i]; }
  return f;
}

__device__ __forceinline__ float gelu_exact(float v) {
  return 0.5f * v * (1.0f + erff(v * 0.7071067811865476f));
}

// ---------------- weight prep: src[b][r][c] f32 -> dst[b][c][r] f16 ----------
__global__ void k_transpose_w(const float* __restrict__ src,
                              _Float16* __restrict__ dst,
                              int batch, int rows, int cols) {
  int idx = blockIdx.x * blockDim.x + threadIdx.x;
  int total = batch * rows * cols;
  if (idx >= total) return;
  int c = idx % cols;
  int r = (idx / cols) % rows;
  int b = idx / (cols * rows);
  dst[((size_t)b * cols + c) * rows + r] = (_Float16)src[idx];
}

// ---------------- stage 1: LN1 + QKV projections -----------------------------
// grid (BB*NT16, HH), block 32. qf/kf: [b][h][n][d]; vt: [b][h][d][n]
__global__ void __launch_bounds__(32) k_ln_qkv(
    const float* __restrict__ x,
    const float* __restrict__ g1, const float* __restrict__ be1,
    const _Float16* __restrict__ wqT, const float* __restrict__ bq,
    const _Float16* __restrict__ wkT, const float* __restrict__ bk,
    const _Float16* __restrict__ wvT, const float* __restrict__ bv,
    _Float16* __restrict__ qf, _Float16* __restrict__ kf,
    _Float16* __restrict__ vt) {
  __shared__ __align__(16) _Float16 h1s[16][72];
  const int lane = threadIdx.x & 31;
  const int h = blockIdx.y;
  const int b = blockIdx.x / NT16;
  const int n0 = (blockIdx.x % NT16) * 16;
  const int row = lane & 15, half = lane >> 4;

  // LayerNorm over head_dim: 2 lanes per row, 32 elems each
  const float* xr = x + ((size_t)(b * NN + n0 + row) * HH + h) * HD + half * 32;
  float vals[32], sum = 0.f, ssq = 0.f;
#pragma unroll
  for (int i = 0; i < 8; ++i) {
    float4 t = ((const float4*)xr)[i];
    vals[4 * i + 0] = t.x; vals[4 * i + 1] = t.y;
    vals[4 * i + 2] = t.z; vals[4 * i + 3] = t.w;
    sum += t.x + t.y + t.z + t.w;
    ssq += t.x * t.x + t.y * t.y + t.z * t.z + t.w * t.w;
  }
  sum += __shfl_xor(sum, 16);
  ssq += __shfl_xor(ssq, 16);
  const float mean = sum * (1.f / 64.f);
  const float rstd = rsqrtf(ssq * (1.f / 64.f) - mean * mean + 1e-5f);
#pragma unroll
  for (int i = 0; i < 32; ++i) {
    int d = half * 32 + i;
    h1s[row][d] = (_Float16)((vals[i] - mean) * rstd * g1[d] + be1[d]);
  }
  __syncthreads();

  const v16h a0 = frag_ld(&h1s[0][0], 72);
  const v16h a1 = frag_ld(&h1s[0][0] + 32, 72);
  const int col = lane & 15, rb = (lane >> 4) * 8;

  {  // Q
    const _Float16* wp = wqT + (size_t)h * HD * HD;
    _Float16* op = qf + ((size_t)(b * HH + h) * NN + n0) * HD;
#pragma unroll
    for (int t = 0; t < 4; ++t) {
      v8f c = {};
      c = wmma_ff(a0, frag_ld(wp + (size_t)(t * 16) * HD, HD), c);
      c = wmma_ff(a1, frag_ld(wp + (size_t)(t * 16) * HD + 32, HD), c);
      float bias = bq[h * HD + t * 16 + col];
#pragma unroll
      for (int r = 0; r < 8; ++r)
        op[(size_t)(rb + r) * HD + t * 16 + col] = (_Float16)(c[r] + bias);
    }
  }
  {  // K
    const _Float16* wp = wkT + (size_t)h * HD * HD;
    _Float16* op = kf + ((size_t)(b * HH + h) * NN + n0) * HD;
#pragma unroll
    for (int t = 0; t < 4; ++t) {
      v8f c = {};
      c = wmma_ff(a0, frag_ld(wp + (size_t)(t * 16) * HD, HD), c);
      c = wmma_ff(a1, frag_ld(wp + (size_t)(t * 16) * HD + 32, HD), c);
      float bias = bk[h * HD + t * 16 + col];
#pragma unroll
      for (int r = 0; r < 8; ++r)
        op[(size_t)(rb + r) * HD + t * 16 + col] = (_Float16)(c[r] + bias);
    }
  }
  {  // V (stored transposed [d][n] so PV B-fragments load row-major)
    const _Float16* wp = wvT + (size_t)h * HD * HD;
    _Float16* op = vt + (size_t)(b * HH + h) * HD * NN;
#pragma unroll
    for (int t = 0; t < 4; ++t) {
      v8f c = {};
      c = wmma_ff(a0, frag_ld(wp + (size_t)(t * 16) * HD, HD), c);
      c = wmma_ff(a1, frag_ld(wp + (size_t)(t * 16) * HD + 32, HD), c);
      float bias = bv[h * HD + t * 16 + col];
#pragma unroll
      for (int r = 0; r < 8; ++r)
        op[(size_t)(t * 16 + col) * NN + (n0 + rb + r)] = (_Float16)(c[r] + bias);
    }
  }
}

// ---------------- stage 2: flash attention per (b,h,16-row Q tile) -----------
// grid (NT16, BB*HH), block 32. attnf: [b][n][h][d] f16
__global__ void __launch_bounds__(32) k_attn(
    const _Float16* __restrict__ qf, const _Float16* __restrict__ kf,
    const _Float16* __restrict__ vt, _Float16* __restrict__ attnf) {
  __shared__ __align__(16) float Sl[16][36];
  __shared__ __align__(16) _Float16 Pl[16][40];
  __shared__ float fac[16];

  const int lane = threadIdx.x & 31;
  const int n0 = blockIdx.x * 16;
  const int bh = blockIdx.y;
  const int b = bh / HH, h = bh % HH;
  const int col = lane & 15, rb = (lane >> 4) * 8;

  const _Float16* qp = qf + ((size_t)bh * NN + n0) * HD;
  const _Float16* kp = kf + (size_t)bh * NN * HD;
  const _Float16* vp = vt + (size_t)bh * HD * NN;

  const v16h qa0 = frag_ld(qp, HD);
  const v16h qa1 = frag_ld(qp + 32, HD);

  v8f zero = {};
  v8f o[4] = {zero, zero, zero, zero};
  float m_r = -1e30f, l_r = 0.f;  // valid on lanes 0..15 (row owners)

  for (int j0 = 0; j0 < NN; j0 += 32) {
    // S tile: 16 rows x 32 keys, Q(16x64) * K^T
    v8f s0 = zero, s1 = zero;
    s0 = wmma_ff(qa0, frag_ld(kp + (size_t)j0 * HD, HD), s0);
    s0 = wmma_ff(qa1, frag_ld(kp + (size_t)j0 * HD + 32, HD), s0);
    s1 = wmma_ff(qa0, frag_ld(kp + (size_t)(j0 + 16) * HD, HD), s1);
    s1 = wmma_ff(qa1, frag_ld(kp + (size_t)(j0 + 16) * HD + 32, HD), s1);
#pragma unroll
    for (int r = 0; r < 8; ++r) {
      Sl[rb + r][col] = s0[r] * SOFT_SCALE;
      Sl[rb + r][16 + col] = s1[r] * SOFT_SCALE;
    }
    __syncthreads();

    if (lane < 16) {  // online softmax, one lane per row
      float tm = -1e30f;
#pragma unroll
      for (int j = 0; j < 32; ++j) tm = fmaxf(tm, Sl[lane][j]);
      float nm = fmaxf(m_r, tm);
      float al = __expf(m_r - nm);
      float ss = 0.f;
#pragma unroll
      for (int j = 0; j < 32; ++j) {
        float p = __expf(Sl[lane][j] - nm);
        Pl[lane][j] = (_Float16)p;
        ss += p;
      }
      l_r = l_r * al + ss;
      m_r = nm;
      fac[lane] = al;
    }
    __syncthreads();

    float af[8];
#pragma unroll
    for (int r = 0; r < 8; ++r) af[r] = fac[rb + r];
#pragma unroll
    for (int t = 0; t < 4; ++t)
#pragma unroll
      for (int r = 0; r < 8; ++r) o[t][r] *= af[r];

    const v16h pa = frag_ld(&Pl[0][0], 40);  // P: 16x32
#pragma unroll
    for (int t = 0; t < 4; ++t) {
      // B[k][n] = V[j0+k][t*16+n] = vt[(t*16+n)*NN + j0+k]
      o[t] = wmma_ff(pa, frag_ld(vp + (size_t)(t * 16) * NN + j0, NN), o[t]);
    }
    __syncthreads();
  }

  if (lane < 16) fac[lane] = 1.f / l_r;
  __syncthreads();
  float inv[8];
#pragma unroll
  for (int r = 0; r < 8; ++r) inv[r] = fac[rb + r];

  _Float16* ap = attnf + ((size_t)(b * NN + n0) * HH + h) * HD;
#pragma unroll
  for (int t = 0; t < 4; ++t)
#pragma unroll
    for (int r = 0; r < 8; ++r)
      ap[(size_t)(rb + r) * (HH * HD) + t * 16 + col] =
          (_Float16)(o[t][r] * inv[r]);
}

// ---------------- stage 3: Wo proj + residual + LN2 + FFN + residual ---------
// grid (BB*NT16, HH), block 32. out2f: [b][n][f] f16
__global__ void __launch_bounds__(32) k_proj_ffn(
    const float* __restrict__ x, const _Float16* __restrict__ attnf,
    const _Float16* __restrict__ woT, const float* __restrict__ bo,
    const float* __restrict__ g2, const float* __restrict__ be2,
    const _Float16* __restrict__ w1T, const float* __restrict__ b1,
    const _Float16* __restrict__ w2T, const float* __restrict__ b2,
    _Float16* __restrict__ out2f) {
  __shared__ __align__(16) float o1[16][68];
  __shared__ __align__(16) _Float16 hs[16][72];
  __shared__ __align__(16) _Float16 t1s[16][264];

  const int lane = threadIdx.x & 31;
  const int h = blockIdx.y;
  const int b = blockIdx.x / NT16;
  const int n0 = (blockIdx.x % NT16) * 16;
  const int col = lane & 15, rb = (lane >> 4) * 8;

  // attn @ wo + bo + x  -> o1 (f32, LDS)
  const _Float16* ap = attnf + ((size_t)(b * NN + n0) * HH + h) * HD;
  const v16h a0 = frag_ld(ap, HH * HD);
  const v16h a1 = frag_ld(ap + 32, HH * HD);
#pragma unroll
  for (int t = 0; t < 4; ++t) {
    v8f c = {};
    const _Float16* wp = woT + ((size_t)h * HD + t * 16) * HD;
    c = wmma_ff(a0, frag_ld(wp, HD), c);
    c = wmma_ff(a1, frag_ld(wp + 32, HD), c);
    float bias = bo[h * HD + t * 16 + col];
#pragma unroll
    for (int r = 0; r < 8; ++r) {
      int rr = rb + r, d = t * 16 + col;
      o1[rr][d] = c[r] + bias +
                  x[((size_t)(b * NN + n0 + rr) * HH + h) * HD + d];
    }
  }
  __syncthreads();

  {  // LN2
    const int row = lane & 15, half = lane >> 4;
    float sum = 0.f, ssq = 0.f;
#pragma unroll
    for (int i = 0; i < 32; ++i) {
      float v = o1[row][half * 32 + i];
      sum += v; ssq += v * v;
    }
    sum += __shfl_xor(sum, 16);
    ssq += __shfl_xor(ssq, 16);
    float mean = sum * (1.f / 64.f);
    float rstd = rsqrtf(ssq * (1.f / 64.f) - mean * mean + 1e-5f);
#pragma unroll
    for (int i = 0; i < 32; ++i) {
      int d = half * 32 + i;
      hs[row][d] = (_Float16)((o1[row][d] - mean) * rstd * g2[d] + be2[d]);
    }
  }
  __syncthreads();

  // FFN1: 64 -> 256, GELU
  const v16h h0 = frag_ld(&hs[0][0], 72);
  const v16h h1 = frag_ld(&hs[0][0] + 32, 72);
#pragma unroll
  for (int t = 0; t < 16; ++t) {
    v8f c = {};
    const _Float16* wp = w1T + ((size_t)h * 256 + t * 16) * HD;
    c = wmma_ff(h0, frag_ld(wp, HD), c);
    c = wmma_ff(h1, frag_ld(wp + 32, HD), c);
    float bias = b1[h * 256 + t * 16 + col];
#pragma unroll
    for (int r = 0; r < 8; ++r)
      t1s[rb + r][t * 16 + col] = (_Float16)gelu_exact(c[r] + bias);
  }
  __syncthreads();

  // FFN2: 256 -> 64, GELU, + residual o1
  v16h tf[8];
#pragma unroll
  for (int kk = 0; kk < 8; ++kk) tf[kk] = frag_ld(&t1s[0][0] + kk * 32, 264);
#pragma unroll
  for (int t = 0; t < 4; ++t) {
    v8f c = {};
    const _Float16* wp = w2T + ((size_t)h * HD + t * 16) * 256;
#pragma unroll
    for (int kk = 0; kk < 8; ++kk)
      c = wmma_ff(tf[kk], frag_ld(wp + kk * 32, 256), c);
    float bias = b2[h * HD + t * 16 + col];
#pragma unroll
    for (int r = 0; r < 8; ++r) {
      int rr = rb + r;
      float u = gelu_exact(c[r] + bias) + o1[rr][t * 16 + col];
      out2f[(size_t)(b * NN + n0 + rr) * FF + h * HD + t * 16 + col] =
          (_Float16)u;
    }
  }
}

// ---------------- stage 4: merge GEMM [4096,1024] @ [1024,1024] + bm ---------
// grid (BB*NT16, FF/64), block 32
__global__ void __launch_bounds__(32) k_merge(
    const _Float16* __restrict__ out2f, const _Float16* __restrict__ wmT,
    const float* __restrict__ bm, float* __restrict__ out) {
  const int lane = threadIdx.x & 31;
  const int m0 = blockIdx.x * 16;
  const int c0 = blockIdx.y * 64;
  const _Float16* apb = out2f + (size_t)m0 * FF;

  v8f zero = {};
  v8f c[4] = {zero, zero, zero, zero};
  for (int k0 = 0; k0 < FF; k0 += 32) {
    if (k0 + 32 < FF)
      __builtin_prefetch((const void*)(apb + k0 + 32), 0, 0);
    v16h a = frag_ld(apb + k0, FF);
#pragma unroll
    for (int t = 0; t < 4; ++t)
      c[t] = wmma_ff(a, frag_ld(wmT + (size_t)(c0 + t * 16) * FF + k0, FF), c[t]);
  }
  const int col = lane & 15, rb = (lane >> 4) * 8;
#pragma unroll
  for (int t = 0; t < 4; ++t) {
    float bias = bm[c0 + t * 16 + col];
#pragma unroll
    for (int r = 0; r < 8; ++r)
      out[(size_t)(m0 + rb + r) * FF + c0 + t * 16 + col] = c[t][r] + bias;
  }
}

// ---------------- host launch ------------------------------------------------
extern "C" void kernel_launch(void* const* d_in, const int* in_sizes, int n_in,
                              void* d_out, int out_size, void* d_ws,
                              size_t ws_size, hipStream_t stream) {
  const float* x   = (const float*)d_in[0];
  const float* g1  = (const float*)d_in[1];
  const float* be1 = (const float*)d_in[2];
  const float* g2  = (const float*)d_in[3];
  const float* be2 = (const float*)d_in[4];
  const float* wq  = (const float*)d_in[5];
  const float* bq  = (const float*)d_in[6];
  const float* wk  = (const float*)d_in[7];
  const float* bk  = (const float*)d_in[8];
  const float* wv  = (const float*)d_in[9];
  const float* bv  = (const float*)d_in[10];
  const float* wo  = (const float*)d_in[11];
  const float* bo  = (const float*)d_in[12];
  const float* w1  = (const float*)d_in[13];
  const float* b1  = (const float*)d_in[14];
  const float* w2  = (const float*)d_in[15];
  const float* b2  = (const float*)d_in[16];
  const float* wm  = (const float*)d_in[17];
  const float* bm  = (const float*)d_in[18];
  float* out = (float*)d_out;

  // workspace carve (f16 buffers), 256B aligned
  char* wsp = (char*)d_ws;
  auto carve = [&](size_t elems) {
    _Float16* p = (_Float16*)wsp;
    wsp += (elems * sizeof(_Float16) + 255) & ~(size_t)255;
    return p;
  };
  _Float16* wqT = carve((size_t)HH * HD * HD);
  _Float16* wkT = carve((size_t)HH * HD * HD);
  _Float16* wvT = carve((size_t)HH * HD * HD);
  _Float16* woT = carve((size_t)HH * HD * HD);
  _Float16* w1T = carve((size_t)HH * HD * 256);
  _Float16* w2T = carve((size_t)HH * 256 * HD);
  _Float16* wmT = carve((size_t)FF * FF);
  _Float16* qf  = carve((size_t)BB * HH * NN * HD);
  _Float16* kf  = carve((size_t)BB * HH * NN * HD);
  _Float16* vt  = carve((size_t)BB * HH * HD * NN);
  _Float16* attnf = carve((size_t)BB * NN * HH * HD);
  _Float16* out2f = carve((size_t)BB * NN * FF);

  auto tlaunch = [&](const float* src, _Float16* dst, int batch, int rows,
                     int cols) {
    int total = batch * rows * cols;
    k_transpose_w<<<(total + 255) / 256, 256, 0, stream>>>(src, dst, batch,
                                                           rows, cols);
  };
  tlaunch(wq, wqT, HH, HD, HD);
  tlaunch(wk, wkT, HH, HD, HD);
  tlaunch(wv, wvT, HH, HD, HD);
  tlaunch(wo, woT, HH, HD, HD);
  tlaunch(w1, w1T, HH, HD, 256);
  tlaunch(w2, w2T, HH, 256, HD);
  tlaunch(wm, wmT, 1, FF, FF);

  dim3 blk(32);
  k_ln_qkv<<<dim3(BB * NT16, HH), blk, 0, stream>>>(
      x, g1, be1, wqT, bq, wkT, bk, wvT, bv, qf, kf, vt);
  k_attn<<<dim3(NT16, BB * HH), blk, 0, stream>>>(qf, kf, vt, attnf);
  k_proj_ffn<<<dim3(BB * NT16, HH), blk, 0, stream>>>(
      x, attnf, woT, bo, g2, be2, w1T, b1, w2T, b2, out2f);
  k_merge<<<dim3(BB * NT16, FF / 64), blk, 0, stream>>>(out2f, wmT, bm, out);
  (void)in_sizes; (void)n_in; (void)out_size; (void)ws_size;
}
